// ImageSelfAttention_78804059947511
// MI455X (gfx1250) — compile-verified, matrix operand order checked by reading, not run on hardware
//
#include <hip/hip_runtime.h>

// ---------------------------------------------------------------------------
// Image self-attention for MI455X (gfx1250, wave32, WMMA bf16 16x16x32).
// B=32, C=256, H=W=32 -> N=1024, D(q/k)=64.
//
// Pipeline:
//   k_cvt_w : wq/wk/wv f32 -> bf16 stacked [384x256]; biases -> bAll[384]
//   k_proj  : qT[b][n][64], kT[b][n][64], v[b][c][n] (all bf16) via WMMA
//   k_stats : colmax/colsum over i for softmax(axis=i) per (b, j)
//   k_attn  : recompute S tiles, P=exp(S-m)/sum in LDS, O = V x P via WMMA,
//             out = x + gamma * O
// ---------------------------------------------------------------------------

typedef __bf16 bf16_t;
typedef __attribute__((ext_vector_type(16))) __bf16 v16bf;
typedef __attribute__((ext_vector_type(8)))  float  v8f;

union FragAB { v16bf v; uint4 q[2]; };
union FragCD { v8f   v; float f[8]; };

__device__ __forceinline__ unsigned short f2bf(float x) {
  union { float f; unsigned u; } c; c.f = x;
  unsigned u = c.u + 0x7FFFu + ((c.u >> 16) & 1u);   // round-to-nearest-even
  return (unsigned short)(u >> 16);
}
__device__ __forceinline__ unsigned pack2(float a, float b) {
  return (unsigned)f2bf(a) | ((unsigned)f2bf(b) << 16);
}

// Load a 16x32 bf16 A/B fragment from a row-major source.
// rowPtr = &M[row][0]; per-lane K range: [kc+8h, +8) and [kc+16+8h, +8).
__device__ __forceinline__ void load_frag(FragAB& f, const bf16_t* rowPtr,
                                          int kc, int half) {
  f.q[0] = *(const uint4*)(rowPtr + kc + 8 * half);
  f.q[1] = *(const uint4*)(rowPtr + kc + 16 + 8 * half);
}

__device__ __forceinline__ v8f wmma_bf16(const FragAB& a, const FragAB& b,
                                         const v8f& c) {
  return __builtin_amdgcn_wmma_f32_16x16x32_bf16(
      false, a.v, false, b.v, (short)0, c, false, false);
}

#define BATCH 32
#define CH    256
#define NPIX  1024
#define DQK   64

// ---------------------------------------------------------------------------
// K0: weights f32 -> bf16 stacked [wq(64x256) | wk(64x256) | wv(256x256)],
//     biases f32 stacked bAll[384] = [bq | bk | bv].
// ---------------------------------------------------------------------------
__global__ __launch_bounds__(256) void k_cvt_w(const float* __restrict__ wq,
                                               const float* __restrict__ wk,
                                               const float* __restrict__ wv_,
                                               const float* __restrict__ bq,
                                               const float* __restrict__ bk,
                                               const float* __restrict__ bv,
                                               unsigned short* __restrict__ wAll,
                                               float* __restrict__ bAll) {
  int i = blockIdx.x * blockDim.x + threadIdx.x;
  if (i < 98304) {
    float v;
    if (i < 16384)      v = wq[i];
    else if (i < 32768) v = wk[i - 16384];
    else                v = wv_[i - 32768];
    wAll[i] = f2bf(v);
  } else if (i < 98688) {
    int j = i - 98304;
    float v;
    if (j < 64)       v = bq[j];
    else if (j < 128) v = bk[j - 64];
    else              v = bv[j - 128];
    bAll[j] = v;
  }
}

// ---------------------------------------------------------------------------
// K1: QKV projection.  Grid (N/16, B), 256 threads (8 waves).
// Each wave owns 3 of 24 m-tiles of the stacked 384x256 weight matrix.
// Stacked layout => tile t base offset is simply t*4096 (branch-free).
// ---------------------------------------------------------------------------
__global__ __launch_bounds__(256)
void k_proj(const float* __restrict__ x, const bf16_t* __restrict__ wAll,
            const float* __restrict__ bAll,
            bf16_t* __restrict__ qT, bf16_t* __restrict__ kT,
            bf16_t* __restrict__ vO) {
  __shared__ unsigned short xs[16 * 264];   // x tile, col-major [n][c], padded

  const int b   = blockIdx.y;
  const int n0  = blockIdx.x * 16;
  const int tid = threadIdx.x;

  // Stage x[b, :, n0:n0+16] -> bf16 LDS (column-major, stride 264).
  const float* xb = x + (size_t)b * CH * NPIX;
  {
    int col = tid & 15, crow = tid >> 4;
#pragma unroll
    for (int it = 0; it < 16; ++it) {
      int c = crow + it * 16;
      xs[col * 264 + c] = f2bf(xb[(size_t)c * NPIX + n0 + col]);
    }
  }
  __syncthreads();

  const int wv   = tid >> 5;
  const int lane = tid & 31;
  const int half = lane >> 4;
  const int lr   = lane & 15;

  FragCD acc[3];
#pragma unroll
  for (int s = 0; s < 3; ++s)
#pragma unroll
    for (int r = 0; r < 8; ++r) acc[s].f[r] = 0.f;

  // Wave's first tile row pointer in the stacked weight matrix; tiles s=0..2
  // are at constant offsets +0 / +4096 / +8192.
  const bf16_t* wrow = wAll + (size_t)(wv * 3) * 4096 + (size_t)lr * 256;
  const unsigned short* xrow = &xs[lr * 264];

#pragma unroll
  for (int kc = 0; kc < 256; kc += 32) {
    FragAB bfrag;   // B = x tile [c, n], lane = column n
    bfrag.q[0] = *(const uint4*)(xrow + kc + 8 * half);
    bfrag.q[1] = *(const uint4*)(xrow + kc + 16 + 8 * half);
    FragAB a0, a1, a2;   // A = weight tiles [o, c], lane = row o
    load_frag(a0, wrow,        kc, half);
    load_frag(a1, wrow + 4096, kc, half);
    load_frag(a2, wrow + 8192, kc, half);
    acc[0].v = wmma_bf16(a0, bfrag, acc[0].v);
    acc[1].v = wmma_bf16(a1, bfrag, acc[1].v);
    acc[2].v = wmma_bf16(a2, bfrag, acc[2].v);
  }

#pragma unroll
  for (int s = 0; s < 3; ++s) {
    const int t   = wv * 3 + s;
    const int og0 = t * 16;               // global output row base (0..368)
#pragma unroll
    for (int r = 0; r < 8; ++r) acc[s].f[r] += bAll[og0 + r + 8 * half];

    if (t < 8) {
      // q/k: transposed store [n][c]: lane writes 8 contiguous bf16 (16B)
      const int o0 = (t < 4) ? og0 : og0 - 64;
      bf16_t* base = (t < 4 ? qT : kT) + (size_t)b * NPIX * DQK +
                     (size_t)(n0 + lr) * DQK + o0 + 8 * half;
      uint4 pkt;
      unsigned* pw = (unsigned*)&pkt;
      pw[0] = pack2(acc[s].f[0], acc[s].f[1]);
      pw[1] = pack2(acc[s].f[2], acc[s].f[3]);
      pw[2] = pack2(acc[s].f[4], acc[s].f[5]);
      pw[3] = pack2(acc[s].f[6], acc[s].f[7]);
      *(uint4*)base = pkt;
    } else {
      // v: natural store v[c][n]
      const int o0 = og0 - 128;
      unsigned short* base =
          (unsigned short*)vO + (size_t)b * CH * NPIX + n0 + lr;
#pragma unroll
      for (int r = 0; r < 8; ++r)
        base[(size_t)(o0 + r + 8 * half) * NPIX] = f2bf(acc[s].f[r]);
    }
  }
}

// ---------------------------------------------------------------------------
// K2: softmax stats over i: colmax[b][j], colsum[b][j].  Grid (N/16, B).
// ---------------------------------------------------------------------------
__global__ __launch_bounds__(256)
void k_stats(const bf16_t* __restrict__ qT, const bf16_t* __restrict__ kT,
             float* __restrict__ colmax, float* __restrict__ colsum) {
  __shared__ float red[8][16];
  __shared__ float cmx[16];

  const int b = blockIdx.y, j0 = blockIdx.x * 16;
  const int tid = threadIdx.x, wv = tid >> 5, lane = tid & 31;
  const int half = lane >> 4, lr = lane & 15;

  const bf16_t* qb = qT + (size_t)b * NPIX * DQK;
  const bf16_t* kb = kT + (size_t)b * NPIX * DQK;

  FragAB kf0, kf1;   // B operand: keys k[:, j], lane = column j
  const bf16_t* krow = kb + (size_t)(j0 + lr) * DQK;
  load_frag(kf0, krow, 0, half);
  load_frag(kf1, krow, 32, half);

  float lmax = -3.0e38f;
#pragma unroll
  for (int it = 0; it < 8; ++it) {
    const bf16_t* qrow = qb + (size_t)((it * 8 + wv) * 16 + lr) * DQK;
    FragCD s;
#pragma unroll
    for (int r = 0; r < 8; ++r) s.f[r] = 0.f;
    FragAB a0, a1;   // A operand: qT rows (i)
    load_frag(a0, qrow, 0, half);
    load_frag(a1, qrow, 32, half);
    s.v = wmma_bf16(a0, kf0, s.v);
    s.v = wmma_bf16(a1, kf1, s.v);
#pragma unroll
    for (int r = 0; r < 8; ++r) lmax = fmaxf(lmax, s.f[r]);
  }
  lmax = fmaxf(lmax, __shfl_xor(lmax, 16, 32));
  if (lane < 16) red[wv][lane] = lmax;
  __syncthreads();
  if (tid < 16) {
    float m = red[0][tid];
#pragma unroll
    for (int w = 1; w < 8; ++w) m = fmaxf(m, red[w][tid]);
    cmx[tid] = m;
  }
  __syncthreads();

  const float cm = cmx[lr];
  float lsum = 0.f;
#pragma unroll
  for (int it = 0; it < 8; ++it) {
    const bf16_t* qrow = qb + (size_t)((it * 8 + wv) * 16 + lr) * DQK;
    FragCD s;
#pragma unroll
    for (int r = 0; r < 8; ++r) s.f[r] = 0.f;
    FragAB a0, a1;
    load_frag(a0, qrow, 0, half);
    load_frag(a1, qrow, 32, half);
    s.v = wmma_bf16(a0, kf0, s.v);
    s.v = wmma_bf16(a1, kf1, s.v);
#pragma unroll
    for (int r = 0; r < 8; ++r) lsum += __expf(s.f[r] - cm);
  }
  lsum += __shfl_xor(lsum, 16, 32);
  if (lane < 16) red[wv][lane] = lsum;
  __syncthreads();
  if (tid < 16) {
    float ssum = 0.f;
#pragma unroll
    for (int w = 0; w < 8; ++w) ssum += red[w][tid];
    colmax[b * NPIX + j0 + tid] = cmx[tid];
    colsum[b * NPIX + j0 + tid] = ssum;
  }
}

// ---------------------------------------------------------------------------
// K3: fused P construction + O = V x P + residual epilogue.  Grid (N/16, B).
// P never leaves LDS.  Each wave owns 2 c-tiles (covers c = 0..255).
// ---------------------------------------------------------------------------
#define PS 136   // padded LDS stride (elements) for P [j][128 i]

__global__ __launch_bounds__(256)
void k_attn(const float* __restrict__ x, const bf16_t* __restrict__ qT,
            const bf16_t* __restrict__ kT, const bf16_t* __restrict__ vO,
            const float* __restrict__ colmax, const float* __restrict__ colsum,
            const float* __restrict__ gamma, float* __restrict__ out) {
  __shared__ unsigned short pl[16 * PS];

  const int b = blockIdx.y, j0 = blockIdx.x * 16;
  const int tid = threadIdx.x, wv = tid >> 5, lane = tid & 31;
  const int half = lane >> 4, lr = lane & 15;

  const bf16_t* qb = qT + (size_t)b * NPIX * DQK;
  const bf16_t* kb = kT + (size_t)b * NPIX * DQK;
  const bf16_t* vb = vO + (size_t)b * CH * NPIX;

  const float cm   = colmax[b * NPIX + j0 + lr];
  const float rinv = 1.0f / colsum[b * NPIX + j0 + lr];

  FragAB kf0, kf1;
  const bf16_t* krow = kb + (size_t)(j0 + lr) * DQK;
  load_frag(kf0, krow, 0, half);
  load_frag(kf1, krow, 32, half);

  const int m0 = (wv * 2) * 16, m1 = (wv * 2 + 1) * 16;
  const bf16_t* vrow0 = vb + (size_t)(m0 + lr) * NPIX;
  const bf16_t* vrow1 = vb + (size_t)(m1 + lr) * NPIX;
  const unsigned short* prow = &pl[lr * PS];

  FragCD oa, ob_;
#pragma unroll
  for (int r = 0; r < 8; ++r) { oa.f[r] = 0.f; ob_.f[r] = 0.f; }

  for (int ib = 0; ib < 8; ++ib) {
    // --- phase A: this wave computes S tile for i-tile (ib*8 + wv) ---
    const bf16_t* qrow = qb + (size_t)((ib * 8 + wv) * 16 + lr) * DQK;
    FragCD s;
#pragma unroll
    for (int r = 0; r < 8; ++r) s.f[r] = 0.f;
    FragAB a0, a1;
    load_frag(a0, qrow, 0, half);
    load_frag(a1, qrow, 32, half);
    s.v = wmma_bf16(a0, kf0, s.v);
    s.v = wmma_bf16(a1, kf1, s.v);

    uint4 pkt;
    unsigned* pw = (unsigned*)&pkt;
    pw[0] = pack2(__expf(s.f[0] - cm) * rinv, __expf(s.f[1] - cm) * rinv);
    pw[1] = pack2(__expf(s.f[2] - cm) * rinv, __expf(s.f[3] - cm) * rinv);
    pw[2] = pack2(__expf(s.f[4] - cm) * rinv, __expf(s.f[5] - cm) * rinv);
    pw[3] = pack2(__expf(s.f[6] - cm) * rinv, __expf(s.f[7] - cm) * rinv);
    // P[j][i_local]: lane (column j = lr) writes its 8 rows i = wv*16+8h+r
    *(uint4*)&pl[lr * PS + wv * 16 + 8 * half] = pkt;
    __syncthreads();

    // --- phase B: O accumulation over this 128-wide i block ---
#pragma unroll
    for (int kc = 0; kc < 128; kc += 32) {
      FragAB pb;   // B operand: P[i, j], lane = column j, K = i contiguous
      pb.q[0] = *(const uint4*)(prow + kc + 8 * half);
      pb.q[1] = *(const uint4*)(prow + kc + 16 + 8 * half);
      FragAB av0, av1;   // A operand: v[c, i], lane = row c
      load_frag(av0, vrow0 + ib * 128, kc, half);
      load_frag(av1, vrow1 + ib * 128, kc, half);
      oa.v  = wmma_bf16(av0, pb, oa.v);
      ob_.v = wmma_bf16(av1, pb, ob_.v);
    }
    __syncthreads();
  }

  // --- epilogue: out = x + gamma * O ---
  const float g = gamma[0];
  const float* xb = x   + (size_t)b * CH * NPIX;
  float*       ob = out + (size_t)b * CH * NPIX;
  const int n = j0 + lr;
#pragma unroll
  for (int r = 0; r < 8; ++r) {
    {
      size_t idx = (size_t)(m0 + r + 8 * half) * NPIX + n;
      ob[idx] = xb[idx] + g * oa.f[r];
    }
    {
      size_t idx = (size_t)(m1 + r + 8 * half) * NPIX + n;
      ob[idx] = xb[idx] + g * ob_.f[r];
    }
  }
}

// ---------------------------------------------------------------------------
// Host launcher
// ---------------------------------------------------------------------------
extern "C" void kernel_launch(void* const* d_in, const int* in_sizes, int n_in,
                              void* d_out, int out_size, void* d_ws,
                              size_t ws_size, hipStream_t stream) {
  (void)in_sizes; (void)n_in; (void)out_size; (void)ws_size;
  const float* x     = (const float*)d_in[0];
  const float* wq    = (const float*)d_in[1];
  const float* bq    = (const float*)d_in[2];
  const float* wk    = (const float*)d_in[3];
  const float* bk    = (const float*)d_in[4];
  const float* wv_   = (const float*)d_in[5];
  const float* bv    = (const float*)d_in[6];
  const float* gamma = (const float*)d_in[7];
  float* out = (float*)d_out;

  char* ws = (char*)d_ws;
  bf16_t* qTb    = (bf16_t*)(ws);                       // 4 MB
  bf16_t* kTb    = (bf16_t*)(ws + 4194304);             // 4 MB
  bf16_t* vObuf  = (bf16_t*)(ws + 8388608);             // 16 MB
  float*  colmax = (float*) (ws + 25165824);            // 128 KB
  float*  colsum = (float*) (ws + 25296896);            // 128 KB
  unsigned short* wAll = (unsigned short*)(ws + 25427968);  // 192 KB
  float*  bAll   = (float*) (ws + 25624576);            // 1.5 KB

  k_cvt_w<<<dim3(386), dim3(256), 0, stream>>>(wq, wk, wv_, bq, bk, bv,
                                               wAll, bAll);
  k_proj<<<dim3(64, 32), dim3(256), 0, stream>>>(
      x, (const bf16_t*)wAll, bAll, qTb, kTb, vObuf);
  k_stats<<<dim3(64, 32), dim3(256), 0, stream>>>(qTb, kTb, colmax, colsum);
  k_attn<<<dim3(64, 32), dim3(256), 0, stream>>>(
      x, qTb, kTb, vObuf, colmax, colsum, gamma, out);
}
